// DuelingGATNet_88613765251554
// MI455X (gfx1250) — matrix operand order repaired and checked by reading, not compiled, not run on hardware
//
#include <hip/hip_runtime.h>
#include <hip/hip_bf16.h>

typedef __attribute__((ext_vector_type(16))) _Float16 v16h;
typedef __attribute__((ext_vector_type(8)))  _Float16 v8h;
typedef __attribute__((ext_vector_type(8)))  float    v8f;

#define N_NODES 32768
#define N_PER   512
#define BGRAPH  64
#define NEDGES  524288
#define TOTAL_E (NEDGES + N_NODES)   // with self loops = 557056
#define C_IN    32
#define HDIM    128
#define KTLS    8
#define A_DIM   8
#define NEG_SLOPE 0.2f

// ---------- monotone float<->uint encoding for atomic max on floats ----------
__device__ __forceinline__ unsigned fenc(float f) {
  unsigned u = __float_as_uint(f);
  return (u & 0x80000000u) ? ~u : (u | 0x80000000u);
}
__device__ __forceinline__ float fdec(unsigned u) {
  u = (u & 0x80000000u) ? (u & 0x7FFFFFFFu) : ~u;
  return __uint_as_float(u);
}

// ---------------------------------------------------------------------------
// Operand packing: f32 -> f16 (same layout), and f32 [K,N] -> f16 [N,K]
// ---------------------------------------------------------------------------
__global__ void convert_f16(const float* __restrict__ src, _Float16* __restrict__ dst,
                            int total) {
  const int i = blockIdx.x * blockDim.x + threadIdx.x;
  if (i < total) dst[i] = (_Float16)src[i];
}

__global__ void transpose_f16(const float* __restrict__ src, _Float16* __restrict__ dst,
                              int K, int N) {   // src[K,N] -> dst[N,K]
  const int i = blockIdx.x * blockDim.x + threadIdx.x;
  if (i >= K * N) return;
  const int k = i / N, n = i % N;
  dst[(size_t)n * K + k] = (_Float16)src[i];
}

// ---------------------------------------------------------------------------
// WMMA GEMM:  C[M,N] = act(A[M,K] * B[K,N] + bias)
//   A  : f16 row-major [M,K]
//   Bt : f16 [N,K]  (B transposed -> per-lane K-contiguous b128 loads)
//   Cf : optional f32 output, Ch : optional f16 output
// One wave computes a 16x32 output tile (2 WMMAs reusing one A fragment).
// M multiple of 16, K multiple of 32, N arbitrary (stores guarded; OOB lanes
// load a clamped valid column -- their accumulator is never stored, and EXEC
// stays all-1s for the WMMAs).
// ---------------------------------------------------------------------------
__global__ void wmma_gemm_f16(const _Float16* __restrict__ A,
                              const _Float16* __restrict__ Bt,
                              const float* __restrict__ bias,
                              float* __restrict__ Cf,
                              _Float16* __restrict__ Ch,
                              int M, int K, int N, int act) {
  const int lane = threadIdx.x & 31;
  const int tile = blockIdx.x * 8 + (threadIdx.x >> 5);
  const int mtiles  = M >> 4;
  const int n2tiles = (N + 31) >> 5;
  if (tile >= mtiles * n2tiles) return;          // wave-uniform
  const int mt = tile / n2tiles;
  const int nt = tile % n2tiles;

  const int l15   = lane & 15;
  const int row_a = mt * 16 + l15;
  const int n0    = nt * 32 + l15;
  const int n1    = n0 + 16;
  const int nc0   = (n0 < N) ? n0 : 0;           // clamp, store-guarded later
  const int nc1   = (n1 < N) ? n1 : 0;
  const int kbA   = (lane < 16) ? 0 : 16;        // A fragment K-base (contiguous layout)
  const int kbB   = (lane < 16) ? 0 : 16;        // B fragment K-base (contiguous layout)

  const _Float16* Ap  = A  + (size_t)row_a * K + kbA;
  const _Float16* Bp0 = Bt + (size_t)nc0  * K + kbB;
  const _Float16* Bp1 = Bt + (size_t)nc1  * K + kbB;

  v8f c0 = {}, c1 = {};
  for (int k0 = 0; k0 < K; k0 += 32) {
    const v8h alo = *(const v8h*)(Ap + k0);        // K = kbA + 0..7
    const v8h ahi = *(const v8h*)(Ap + k0 + 8);    // K = kbA + 8..15
    const v16h a  = __builtin_shufflevector(alo, ahi,
                      0,1,2,3,4,5,6,7,8,9,10,11,12,13,14,15);
    const v8h b0lo = *(const v8h*)(Bp0 + k0);      // K = kbB + 0..7
    const v8h b0hi = *(const v8h*)(Bp0 + k0 + 8);  // K = kbB + 8..15
    const v16h b0  = __builtin_shufflevector(b0lo, b0hi,
                      0,1,2,3,4,5,6,7,8,9,10,11,12,13,14,15);
    const v8h b1lo = *(const v8h*)(Bp1 + k0);
    const v8h b1hi = *(const v8h*)(Bp1 + k0 + 8);
    const v16h b1  = __builtin_shufflevector(b1lo, b1hi,
                      0,1,2,3,4,5,6,7,8,9,10,11,12,13,14,15);
    c0 = __builtin_amdgcn_wmma_f32_16x16x32_f16(false, a, false, b0,
                                                (short)0, c0, false, false);
    c1 = __builtin_amdgcn_wmma_f32_16x16x32_f16(false, a, false, b1,
                                                (short)0, c1, false, false);
  }

  const int rbase = mt * 16 + ((lane < 16) ? 0 : 8);
  if (n0 < N) {
    const float bv = bias ? bias[n0] : 0.f;
#pragma unroll
    for (int v = 0; v < 8; ++v) {
      float val = c0[v] + bv;
      if (act) val = fmaxf(val, 0.f);
      const size_t idx = (size_t)(rbase + v) * N + n0;
      if (Cf) Cf[idx] = val;
      if (Ch) Ch[idx] = (_Float16)val;
    }
  }
  if (n1 < N) {
    const float bv = bias ? bias[n1] : 0.f;
#pragma unroll
    for (int v = 0; v < 8; ++v) {
      float val = c1[v] + bv;
      if (act) val = fmaxf(val, 0.f);
      const size_t idx = (size_t)(rbase + v) * N + n1;
      if (Cf) Cf[idx] = val;
      if (Ch) Ch[idx] = (_Float16)val;
    }
  }
}

// ---------------------------------------------------------------------------
// Per-node attention logit pieces: s = h . att_src, d = h . att_dst
// ---------------------------------------------------------------------------
__global__ void node_dots(const float* __restrict__ h,
                          const float* __restrict__ att_src,
                          const float* __restrict__ att_dst,
                          float* __restrict__ s, float* __restrict__ d) {
  const int lane = threadIdx.x & 31;
  const int node = blockIdx.x * 8 + (threadIdx.x >> 5);
  if (node >= N_NODES) return;
  const float4 hv = reinterpret_cast<const float4*>(h + (size_t)node * HDIM)[lane];
  const float4 as = reinterpret_cast<const float4*>(att_src)[lane];
  const float4 ad = reinterpret_cast<const float4*>(att_dst)[lane];
  float ss = hv.x * as.x + hv.y * as.y + hv.z * as.z + hv.w * as.w;
  float dd = hv.x * ad.x + hv.y * ad.y + hv.z * ad.z + hv.w * ad.w;
#pragma unroll
  for (int off = 16; off > 0; off >>= 1) {
    ss += __shfl_xor(ss, off, 32);
    dd += __shfl_xor(dd, off, 32);
  }
  if (lane == 0) { s[node] = ss; d[node] = dd; }
}

// ---------------------------------------------------------------------------
__global__ void init_buffers(float* __restrict__ outagg,
                             unsigned* __restrict__ m_enc,
                             float* __restrict__ denom) {
  const size_t i = (size_t)blockIdx.x * blockDim.x + threadIdx.x;
  if (i < (size_t)N_NODES * HDIM) outagg[i] = 0.f;
  if (i < N_NODES) { m_enc[i] = 0u; denom[i] = 0.f; }  // 0 < fenc of any logit
}

// Pass 1: per-destination running max of LeakyReLU(s[src] + d[dst])
__global__ void edge_max(const int* __restrict__ ei,
                         const float* __restrict__ s,
                         const float* __restrict__ d,
                         unsigned* __restrict__ m_enc) {
  const int i = blockIdx.x * blockDim.x + threadIdx.x;
  if (i >= TOTAL_E) return;
  int src, dst;
  if (i < NEDGES) { src = ei[i]; dst = ei[NEDGES + i]; }
  else            { src = dst = i - NEDGES; }
  float e = s[src] + d[dst];
  e = (e > 0.f) ? e : NEG_SLOPE * e;
  atomicMax(&m_enc[dst], fenc(e));
}

// Pass 2 (fused): ex = exp(e - m[dst]);  denom[dst] += ex;  outagg[dst,:] += ex * h[src,:]
__global__ void edge_accum(const int* __restrict__ ei,
                           const float* __restrict__ s,
                           const float* __restrict__ d,
                           const unsigned* __restrict__ m_enc,
                           const float* __restrict__ h,
                           float* __restrict__ denom,
                           float* __restrict__ outagg) {
  const int lane = threadIdx.x & 31;
  const int eid  = blockIdx.x * 8 + (threadIdx.x >> 5);
  if (eid >= TOTAL_E) return;
  int src, dst;
  if (eid < NEDGES) { src = ei[eid]; dst = ei[NEDGES + eid]; }
  else              { src = dst = eid - NEDGES; }
  const float* hr = h + (size_t)src * HDIM;
  __builtin_prefetch(hr + lane * 4, 0, 3);     // global_prefetch_b8: hide h-row fetch
  float e = s[src] + d[dst];
  e = (e > 0.f) ? e : NEG_SLOPE * e;
  const float ex = __expf(e - fdec(m_enc[dst]));
  if (lane == 0) atomicAdd(&denom[dst], ex);
  const float4 hv = reinterpret_cast<const float4*>(hr)[lane];
  float* orow = outagg + (size_t)dst * HDIM + lane * 4;
  atomicAdd(orow + 0, ex * hv.x);
  atomicAdd(orow + 1, ex * hv.y);
  atomicAdd(orow + 2, ex * hv.z);
  atomicAdd(orow + 3, ex * hv.w);
}

// Pass 3 (fused): normalize + bias + ELU (in place)
__global__ void finalize_gat(float* __restrict__ outagg,
                             const float* __restrict__ denom,
                             const float* __restrict__ b_gat) {
  const size_t i = (size_t)blockIdx.x * blockDim.x + threadIdx.x;
  if (i >= (size_t)N_NODES * HDIM) return;
  const int node = (int)(i >> 7), c = (int)(i & 127);
  float v = outagg[i] / (denom[node] + 1e-16f) + b_gat[c];
  outagg[i] = (v > 0.f) ? v : (__expf(v) - 1.f);
}

// Gather TLS node features straight into the f16 GEMM-A layout: feats16[B, K*H]
__global__ void gather_feats(const float* __restrict__ gat,
                             const int* __restrict__ tls_idx,
                             _Float16* __restrict__ feats16) {
  const int i = blockIdx.x * blockDim.x + threadIdx.x;   // 64*8*128 = 65536
  const int g = i >> 10, rem = i & 1023;
  const int k = rem >> 7, c = rem & 127;
  const int node = g * N_PER + tls_idx[k];
  feats16[i] = (_Float16)gat[(size_t)node * HDIM + c];
}

// q = v + a - mean(a)
__global__ void combine_heads(const float* __restrict__ v4,
                              const float* __restrict__ a4,
                              float* __restrict__ out) {
  const int b = threadIdx.x;
  if (b >= BGRAPH) return;
  const float vv = v4[b];
  float av[A_DIM]; float mean = 0.f;
#pragma unroll
  for (int j = 0; j < A_DIM; ++j) { av[j] = a4[b * A_DIM + j]; mean += av[j]; }
  mean *= (1.f / A_DIM);
#pragma unroll
  for (int j = 0; j < A_DIM; ++j) out[b * A_DIM + j] = vv + av[j] - mean;
}

// ---------------------------------------------------------------------------
extern "C" void kernel_launch(void* const* d_in, const int* in_sizes, int n_in,
                              void* d_out, int out_size, void* d_ws, size_t ws_size,
                              hipStream_t stream) {
  const float* x       = (const float*)d_in[0];
  const int*   ei      = (const int*)  d_in[1];
  const int*   tls_idx = (const int*)  d_in[2];
  const float* W       = (const float*)d_in[3];
  const float* att_src = (const float*)d_in[4];
  const float* att_dst = (const float*)d_in[5];
  const float* b_gat   = (const float*)d_in[6];
  const float* vW[4] = {(const float*)d_in[7],  (const float*)d_in[9],
                        (const float*)d_in[11], (const float*)d_in[13]};
  const float* vb[4] = {(const float*)d_in[8],  (const float*)d_in[10],
                        (const float*)d_in[12], (const float*)d_in[14]};
  const float* aW[4] = {(const float*)d_in[15], (const float*)d_in[17],
                        (const float*)d_in[19], (const float*)d_in[21]};
  const float* ab[4] = {(const float*)d_in[16], (const float*)d_in[18],
                        (const float*)d_in[20], (const float*)d_in[22]};
  float* out = (float*)d_out;

  // -------- workspace carve-up (256B aligned regions) --------
  char* p = (char*)d_ws;
  auto alloc = [&](size_t bytes) -> void* {
    void* r = (void*)p;
    p += (bytes + 255) & ~(size_t)255;
    return r;
  };
  float*     h      = (float*)alloc((size_t)N_NODES * HDIM * 4);
  float*     outagg = (float*)alloc((size_t)N_NODES * HDIM * 4);
  float*     sbuf   = (float*)alloc(N_NODES * 4);
  float*     dbuf   = (float*)alloc(N_NODES * 4);
  unsigned*  m_enc  = (unsigned*)alloc(N_NODES * 4);
  float*     denom  = (float*)alloc(N_NODES * 4);
  _Float16*  x16    = (_Float16*)alloc((size_t)N_NODES * C_IN * 2);
  _Float16*  Wt     = (_Float16*)alloc((size_t)HDIM * C_IN * 2);         // [128,32]
  _Float16*  feats16= (_Float16*)alloc((size_t)BGRAPH * 1024 * 2);
  // MLP weights, transposed f16 [N,K]
  const int vKs[4] = {1024, 1024, 256, 64}, vNs[4] = {1024, 256, 64, 1};
  const int aNs[4] = {1024, 256, 64, A_DIM};
  _Float16* vWt[4]; _Float16* aWt[4];
  for (int l = 0; l < 4; ++l) {
    vWt[l] = (_Float16*)alloc((size_t)vKs[l] * vNs[l] * 2);
    aWt[l] = (_Float16*)alloc((size_t)vKs[l] * aNs[l] * 2);
  }
  // f16 activations (v-head / a-head); final layer outputs in f32
  _Float16* vact[3]; _Float16* aact[3];
  for (int l = 0; l < 3; ++l) {
    vact[l] = (_Float16*)alloc((size_t)BGRAPH * vNs[l] * 2);
    aact[l] = (_Float16*)alloc((size_t)BGRAPH * aNs[l] * 2);
  }
  float* v4f = (float*)alloc(BGRAPH * 1 * 4);
  float* a4f = (float*)alloc(BGRAPH * A_DIM * 4);

  const dim3 blk(256);
  auto gemm_blocks = [](int M, int N) {
    const int tiles = (M >> 4) * ((N + 31) >> 5);
    return (tiles + 7) / 8;
  };

  // 0) pack operands to f16 (x same-layout; weights transposed [N,K])
  convert_f16<<<(N_NODES * C_IN + 255) / 256, blk, 0, stream>>>(x, x16, N_NODES * C_IN);
  transpose_f16<<<(C_IN * HDIM + 255) / 256, blk, 0, stream>>>(W, Wt, C_IN, HDIM);
  for (int l = 0; l < 4; ++l) {
    transpose_f16<<<(vKs[l] * vNs[l] + 255) / 256, blk, 0, stream>>>(vW[l], vWt[l], vKs[l], vNs[l]);
    transpose_f16<<<(vKs[l] * aNs[l] + 255) / 256, blk, 0, stream>>>(aW[l], aWt[l], vKs[l], aNs[l]);
  }

  // 1) h = x @ W   (M=32768, K=32, N=128; f32 out for attention/scatter)
  wmma_gemm_f16<<<gemm_blocks(N_NODES, HDIM), blk, 0, stream>>>(
      x16, Wt, nullptr, h, nullptr, N_NODES, C_IN, HDIM, 0);
  // 2) per-node attention dot products
  node_dots<<<N_NODES / 8, blk, 0, stream>>>(h, att_src, att_dst, sbuf, dbuf);
  // 3) init segment-softmax state
  init_buffers<<<(N_NODES * HDIM) / 256, blk, 0, stream>>>(outagg, m_enc, denom);
  // 4) pass 1: segment max
  edge_max<<<(TOTAL_E + 255) / 256, blk, 0, stream>>>(ei, sbuf, dbuf, m_enc);
  // 5) pass 2: exp + denom + weighted scatter (one wave per edge)
  edge_accum<<<(TOTAL_E + 7) / 8, blk, 0, stream>>>(ei, sbuf, dbuf, m_enc, h,
                                                    denom, outagg);
  // 6) pass 3: normalize + bias + ELU (in place)
  finalize_gat<<<(N_NODES * HDIM) / 256, blk, 0, stream>>>(outagg, denom, b_gat);
  // 7) gather TLS features, directly into f16 GEMM-A layout [64, 1024]
  gather_feats<<<(BGRAPH * KTLS * HDIM) / 256, blk, 0, stream>>>(outagg, tls_idx, feats16);

  // 8) dueling MLP heads: all WMMA, f16 activations chained in row-major layout
  // value head: 1024 -> 1024 -> 256 -> 64 -> 1
  wmma_gemm_f16<<<gemm_blocks(BGRAPH, 1024), blk, 0, stream>>>(
      feats16, vWt[0], vb[0], nullptr, vact[0], BGRAPH, 1024, 1024, 1);
  wmma_gemm_f16<<<gemm_blocks(BGRAPH, 256), blk, 0, stream>>>(
      vact[0], vWt[1], vb[1], nullptr, vact[1], BGRAPH, 1024, 256, 1);
  wmma_gemm_f16<<<gemm_blocks(BGRAPH, 64), blk, 0, stream>>>(
      vact[1], vWt[2], vb[2], nullptr, vact[2], BGRAPH, 256, 64, 1);
  wmma_gemm_f16<<<gemm_blocks(BGRAPH, 1), blk, 0, stream>>>(
      vact[2], vWt[3], vb[3], v4f, nullptr, BGRAPH, 64, 1, 0);
  // advantage head: 1024 -> 1024 -> 256 -> 64 -> 8
  wmma_gemm_f16<<<gemm_blocks(BGRAPH, 1024), blk, 0, stream>>>(
      feats16, aWt[0], ab[0], nullptr, aact[0], BGRAPH, 1024, 1024, 1);
  wmma_gemm_f16<<<gemm_blocks(BGRAPH, 256), blk, 0, stream>>>(
      aact[0], aWt[1], ab[1], nullptr, aact[1], BGRAPH, 1024, 256, 1);
  wmma_gemm_f16<<<gemm_blocks(BGRAPH, 64), blk, 0, stream>>>(
      aact[1], aWt[2], ab[2], nullptr, aact[2], BGRAPH, 256, 64, 1);
  wmma_gemm_f16<<<gemm_blocks(BGRAPH, A_DIM), blk, 0, stream>>>(
      aact[2], aWt[3], ab[3], a4f, nullptr, BGRAPH, 64, A_DIM, 0);

  // 9) q = v + a - mean(a)
  combine_heads<<<1, 64, 0, stream>>>(v4f, a4f, out);
}
